// NEmbedding_79688823210847
// MI455X (gfx1250) — matrix operand order repaired and checked by reading, not compiled, not run on hardware
//
#include <hip/hip_runtime.h>

typedef __attribute__((ext_vector_type(16))) __bf16 v16bf;
typedef __attribute__((ext_vector_type(8)))  float  v8f;

namespace {
constexpr int kB  = 32768;  // batch rows
constexpr int kF  = 64;     // features
constexpr int kNB = 64;     // bins per feature
constexpr int kE  = 32;     // embedding dim
}

// One block per (M-chunk, feature f). 8 waves/block, each wave computes 4
// M-tiles of 16 rows: out[16x32] = relu( enc[16x64] @ W_f[64x32] + b_f ).
// enc is built directly in WMMA A-layout registers; W_f is staged in LDS in
// WMMA B-layout as a bf16 hi/lo split for ~fp32 accuracy via 3 WMMAs/product.
// Output (256 MB, write-once) is stored non-temporally to keep L2 for x/W.
__global__ __launch_bounds__(256)
void nembed_ple_wmma(const float* __restrict__ xg,
                     const float* __restrict__ binsg,
                     const float* __restrict__ Wg,
                     const float* __restrict__ bg,
                     float* __restrict__ outg)
{
  const int f   = blockIdx.y;
  const int tid = threadIdx.x;

  // B-layout tiles: [tile = kt*2+nt][lane][16 halves] -> 32 contiguous bytes/lane
  __shared__ __align__(16) __bf16 WH[4][32][16];
  __shared__ __align__(16) __bf16 WL[4][32][16];
  __shared__ float bins_s[65];
  __shared__ float invw_s[64];
  __shared__ float bias_s[32];

  // ---- Stage W[f] into LDS, swizzled to WMMA 32x16 B layout, bf16 hi+lo ----
  // B order used here (16-bit, 32x16): lanes 0-15 hold K=0..15 (half j -> K=j),
  // lanes 16-31 hold K=16..31 (half j -> K=16+j); column N = lane%16.
  for (int i = tid; i < 2048; i += 256) {
    const int t  = i >> 9;          // tile 0..3  (kt*2 + nt)
    const int kt = t >> 1;
    const int nt = t & 1;
    const int l  = (i >> 4) & 31;   // lane slot
    const int j  = i & 15;          // half index within lane
    const int K  = kt * 32 + j + 16 * (l >> 4);
    const int N  = nt * 16 + (l & 15);
    const float w  = Wg[(f * kNB + K) * kE + N];
    const __bf16 h = (__bf16)w;
    WH[t][l][j] = h;
    WL[t][l][j] = (__bf16)(w - (float)h);
  }

  // ---- bins, reciprocal widths, bias ----
  if (tid < 64) {
    const float lo = binsg[f * kNB + tid];
    const float hi = (tid < 63) ? binsg[f * kNB + tid + 1] : -1.0f;
    bins_s[tid] = lo;
    invw_s[tid] = 1.0f / (hi - lo);
  }
  if (tid == 0)  bins_s[64] = -1.0f;
  if (tid < 32)  bias_s[tid] = bg[f * kE + tid];
  __syncthreads();

  const int wave = tid >> 5;
  const int lane = tid & 31;
  const int m    = lane & 15;   // row within tile (A) / column (B,C)
  const int lh   = lane >> 4;   // lane half select

  // ---- Pull B tiles from LDS into registers once (reused for all M-tiles) --
  v16bf bh[2][2], bl[2][2];
  #pragma unroll
  for (int kt = 0; kt < 2; ++kt) {
    #pragma unroll
    for (int nt = 0; nt < 2; ++nt) {
      bh[kt][nt] = *(const v16bf*)(&WH[kt * 2 + nt][lane][0]);
      bl[kt][nt] = *(const v16bf*)(&WL[kt * 2 + nt][lane][0]);
    }
  }
  const float bias0 = bias_s[m];
  const float bias1 = bias_s[16 + m];

  // ---- Preload x for all 4 M-tiles so the loads overlap the enc build -----
  float xvs[4];
  #pragma unroll
  for (int it = 0; it < 4; ++it) {
    const int tile = blockIdx.x * 8 + wave + it * 512;  // 0..2047
    xvs[it] = xg[(tile * 16 + m) * kF + f];
  }

  #pragma unroll
  for (int it = 0; it < 4; ++it) {
    const int tile = blockIdx.x * 8 + wave + it * 512;
    const int row0 = tile * 16;
    const float xv = xvs[it];

    // ---- Build enc tile directly in WMMA 16x32 A layout (hi/lo bf16) ------
    // A order used here (16-bit, 16x32): lane half owns K 0..15 / 16..31;
    // half j -> K = j + 16*lh (matches B staging).
    v16bf aH[2], aL[2];
    #pragma unroll
    for (int kt = 0; kt < 2; ++kt) {
      #pragma unroll
      for (int j = 0; j < 16; ++j) {
        const int k = kt * 32 + j + 16 * lh;
        const float lo = bins_s[k];
        const float hi = bins_s[k + 1];           // bins_s[64] == -1.0f
        const bool left  = (xv <  lo) && (k > 0);
        const bool right = (xv >= hi) && (k < kNB - 1);
        const float v = (xv - lo) * invw_s[k];
        const float e = (left != right) ? v : (left ? 0.0f : 1.0f);
        const __bf16 h = (__bf16)e;
        aH[kt][j] = h;
        aL[kt][j] = (__bf16)(e - (float)h);
      }
    }

    // ---- 3-term bf16-split GEMM accumulate, bias preloaded in C -----------
    #pragma unroll
    for (int nt = 0; nt < 2; ++nt) {
      const float bv = (nt == 0) ? bias0 : bias1;
      v8f acc;
      #pragma unroll
      for (int r = 0; r < 8; ++r) acc[r] = bv;
      #pragma unroll
      for (int kt = 0; kt < 2; ++kt) {
        acc = __builtin_amdgcn_wmma_f32_16x16x32_bf16(
                  false, aL[kt], false, bh[kt][nt], (short)0, acc, false, false);
        acc = __builtin_amdgcn_wmma_f32_16x16x32_bf16(
                  false, aH[kt], false, bl[kt][nt], (short)0, acc, false, false);
        acc = __builtin_amdgcn_wmma_f32_16x16x32_bf16(
                  false, aH[kt], false, bh[kt][nt], (short)0, acc, false, false);
      }

      // ---- ReLU + NT store. C layout: VGPR r -> row r + 8*lh, col = lane%16
      const int col = nt * 16 + m;
      #pragma unroll
      for (int r = 0; r < 8; ++r) {
        const int rowc = row0 + r + 8 * lh;
        const float o = acc[r];
        __builtin_nontemporal_store((o > 0.0f) ? o : 0.0f,
                                    &outg[rowc * (kF * kE) + f * kE + col]);
      }
    }
  }
}

extern "C" void kernel_launch(void* const* d_in, const int* in_sizes, int n_in,
                              void* d_out, int out_size, void* d_ws, size_t ws_size,
                              hipStream_t stream) {
  (void)in_sizes; (void)n_in; (void)d_ws; (void)ws_size; (void)out_size;
  const float* x    = (const float*)d_in[0];
  const float* bins = (const float*)d_in[1];
  const float* W    = (const float*)d_in[2];
  const float* b    = (const float*)d_in[3];
  float* out        = (float*)d_out;

  dim3 grid(64, 64, 1);   // grid.x * 8 waves * 4 iters = 2048 M-tiles of 16 rows
  nembed_ple_wmma<<<grid, dim3(256, 1, 1), 0, stream>>>(x, bins, W, b, out);
}
